// PointNetFeaturePropagation_446676598868
// MI455X (gfx1250) — matrix operand order repaired and checked by reading, not compiled, not run on hardware
//
#include <hip/hip_runtime.h>

// ---------------------------------------------------------------------------
// PointNet Feature Propagation for MI455X (gfx1250, wave32, WMMA + async LDS).
// B=4, N=16384, S=2048, D1=128, D2=256, MLP 384->256->128 (BN folded + ReLU).
// GEMMs on v_wmma_f32_16x16x32_f16; W0 staged to LDS with
// global_load_async_to_lds_b128 (ASYNCcnt), double buffered; B fragments are
// software-pipelined (depth 2) so waits are partial, not full drains.
// ---------------------------------------------------------------------------

typedef __attribute__((ext_vector_type(16))) _Float16 v16h;
typedef __attribute__((ext_vector_type(8)))  _Float16 v8h;
typedef __attribute__((ext_vector_type(8)))  float    v8f;
typedef __attribute__((ext_vector_type(4)))  float    f4;

#define BATCH 4
#define NPTS  16384
#define SPTS  2048
#define CD1   128
#define CD2   256
#define INCH  384      // D1 + D2
#define OUT0  256
#define OUT1  128
#define NK0   (INCH / 32)   // 12 k-steps, layer 0
#define NK1   (OUT0 / 32)   // 8  k-steps, layer 1

// workspace byte offsets (total ~50.6 MB)
static const size_t OFF_XH  = 0;                       // 65536*384 halves = 50,331,648 B
static const size_t OFF_W0P = 50331648;                // 256*384 halves (frag-packed)
static const size_t OFF_W1P = OFF_W0P + 196608;        // 128*256 halves (frag-packed)
static const size_t OFF_S0  = OFF_W1P + 65536;
static const size_t OFF_T0  = OFF_S0 + 1024;
static const size_t OFF_S1  = OFF_T0 + 1024;
static const size_t OFF_T1  = OFF_S1 + 512;

// --- CDNA5 async global->LDS copy (16 B per lane per issue, ASYNCcnt) ------
__device__ __forceinline__ void async_b128_to_lds(unsigned lds_byte_off,
                                                  const void* gptr) {
  asm volatile("global_load_async_to_lds_b128 %0, %1, off"
               :: "v"(lds_byte_off), "v"((unsigned long long)(size_t)gptr)
               : "memory");
}
#define WAIT_ASYNC(n) asm volatile("s_wait_asynccnt %0" :: "i"(n) : "memory")

// ---------------------------------------------------------------------------
// Kernel 0: fold BN params; convert + FRAGMENT-PACK weights to f16.
// Packed layout: frag(ks,nt) is a contiguous 1 KB block of 32 lanes x 16
// halves; lane l holds column o = nt*16 + (l&15), K = ks*32 + 16*(j>>3) + 8*(l>>4) + (j&7).
// ---------------------------------------------------------------------------
__global__ __launch_bounds__(256) void fp_prep_kernel(
    const float* __restrict__ w0, const float* __restrict__ b0,
    const float* __restrict__ g0, const float* __restrict__ bt0,
    const float* __restrict__ rm0, const float* __restrict__ rv0,
    const float* __restrict__ w1, const float* __restrict__ b1,
    const float* __restrict__ g1, const float* __restrict__ bt1,
    const float* __restrict__ rm1, const float* __restrict__ rv1,
    _Float16* __restrict__ w0p, _Float16* __restrict__ w1p,
    float* __restrict__ s0, float* __restrict__ t0,
    float* __restrict__ s1, float* __restrict__ t1) {
  const int i = blockIdx.x * 256 + threadIdx.x;
  if (i < OUT0 * INCH) {
    const int j  = i & 15;
    const int l  = (i >> 4) & 31;
    const int fr = i >> 9;         // 0..191
    const int nt = fr & 15;
    const int ks = fr >> 4;        // 0..11
    const int o  = nt * 16 + (l & 15);
    const int k  = ks * 32 + ((j >> 3) << 4) + ((l >> 4) << 3) + (j & 7);
    w0p[i] = (_Float16)w0[o * INCH + k];
  }
  if (i < OUT1 * OUT0) {
    const int j  = i & 15;
    const int l  = (i >> 4) & 31;
    const int fr = i >> 9;         // 0..63
    const int nt = fr & 7;
    const int ks = fr >> 3;        // 0..7
    const int o  = nt * 16 + (l & 15);
    const int k  = ks * 32 + ((j >> 3) << 4) + ((l >> 4) << 3) + (j & 7);
    w1p[i] = (_Float16)w1[o * OUT0 + k];
  }
  if (i < OUT0) {
    float s = g0[i] * rsqrtf(rv0[i] + 1e-5f);
    s0[i] = s;
    t0[i] = (b0[i] - rm0[i]) * s + bt0[i];
  }
  if (i < OUT1) {
    float s = g1[i] * rsqrtf(rv1[i] + 1e-5f);
    s1[i] = s;
    t1[i] = (b1[i] - rm1[i]) * s + bt1[i];
  }
}

// ---------------------------------------------------------------------------
// Kernel 1: 3-NN + inverse-distance interpolation -> f16 feature rows.
// ---------------------------------------------------------------------------
__global__ __launch_bounds__(256) void fp_interp_kernel(
    const float* __restrict__ xyz1,     // [B,N,3]
    const float* __restrict__ xyz2,     // [B,3,S]
    const float* __restrict__ points1,  // [B,N,D1]
    const float* __restrict__ points2,  // [B,D2,S]
    _Float16* __restrict__ Xh) {        // [B*N, 384] f16
  __shared__ float sx[SPTS], sy[SPTS], sz[SPTS];
  const int tid = threadIdx.x;
  const int row = blockIdx.x * 256 + tid;   // row = b*N + n
  const int b   = row >> 14;

  const float* q = xyz2 + (size_t)b * 3 * SPTS;
  for (int i = tid; i < SPTS; i += 256) {
    sx[i] = q[i];
    sy[i] = q[SPTS + i];
    sz[i] = q[2 * SPTS + i];
  }
  __syncthreads();

  const float px = xyz1[(size_t)row * 3 + 0];
  const float py = xyz1[(size_t)row * 3 + 1];
  const float pz = xyz1[(size_t)row * 3 + 2];

  float d0 = 3.0e38f, d1 = 3.0e38f, d2 = 3.0e38f;
  int   i0 = 0, i1 = 0, i2 = 0;
  for (int s = 0; s < SPTS; ++s) {
    const float dx = px - sx[s];
    const float dy = py - sy[s];
    const float dz = pz - sz[s];
    const float d  = fmaf(dx, dx, fmaf(dy, dy, dz * dz));
    if (d < d2) {
      if (d < d1) {
        d2 = d1; i2 = i1;
        if (d < d0) { d1 = d0; i1 = i0; d0 = d; i0 = s; }
        else        { d1 = d;  i1 = s; }
      } else {
        d2 = d; i2 = s;
      }
    }
  }

  const float r0 = 1.0f / (d0 + 1e-8f);
  const float r1 = 1.0f / (d1 + 1e-8f);
  const float r2 = 1.0f / (d2 + 1e-8f);
  const float inv = 1.0f / (r0 + r1 + r2);
  const float wa = r0 * inv, wb = r1 * inv, wc = r2 * inv;

  union PK { _Float16 h[2]; unsigned int u; };

  const float* p1r = points1 + (size_t)row * CD1;
  _Float16* xr = Xh + (size_t)row * INCH;
  for (int c = 0; c < CD1; c += 2) {
    PK pk;
    pk.h[0] = (_Float16)p1r[c];
    pk.h[1] = (_Float16)p1r[c + 1];
    *(unsigned int*)(xr + c) = pk.u;
  }

  const float* p2b = points2 + (size_t)b * CD2 * SPTS;  // [D2][S]
  for (int c = 0; c < CD2; c += 2) {
    const float* pc0 = p2b + (size_t)c * SPTS;
    const float* pc1 = pc0 + SPTS;
    const float v0 = wa * pc0[i0] + wb * pc0[i1] + wc * pc0[i2];
    const float v1 = wa * pc1[i0] + wb * pc1[i1] + wc * pc1[i2];
    PK pk;
    pk.h[0] = (_Float16)v0;
    pk.h[1] = (_Float16)v1;
    *(unsigned int*)(xr + CD1 + c) = pk.u;
  }
}

// ---------------------------------------------------------------------------
// Kernel 2: fused 2-layer MLP on WMMA with async-LDS W0 staging.
// 128 threads = 4 waves; each wave owns 16 rows (block = 64 rows).
// smem (33,792 B) is time-shared: [W0 slice double buffer 2x16KB] then
// [Y0 staging 4x16x264 f16].
// ---------------------------------------------------------------------------
#define Y0_STRIDE 264  // halves; 528 B rows: 16B-aligned

__global__ __launch_bounds__(128) void fp_mlp_kernel(
    const _Float16* __restrict__ Xh,
    const _Float16* __restrict__ w0p,  // frag-packed, 12*16 KB slices
    const _Float16* __restrict__ w1p,  // frag-packed, 8*8 KB slices
    const float* __restrict__ s0g, const float* __restrict__ t0g,
    const float* __restrict__ s1g, const float* __restrict__ t1g,
    float* __restrict__ out) {         // [B,128,N]
  __shared__ __align__(16) char smem[33792];
  _Float16* sh = (_Float16*)smem;

  const int tid  = threadIdx.x;
  const int wave = tid >> 5;
  const int lane = tid & 31;
  const int hi   = lane >> 4;
  const int ln   = lane & 15;
  const int rowBase = blockIdx.x * 64 + wave * 16;

  // LDS byte offset of smem (generic LDS pointers carry the offset in the
  // low 32 bits of the flat address).
  const unsigned lds_base = (unsigned)(size_t)(void*)smem;

  // ---- issue one 16 KB W0 slice copy: 8 async b128 per thread ----
  auto issue_slice = [&](int ks, int buf) {
    const char* gsrc = (const char*)w0p + (size_t)ks * 16384 + tid * 128;
    const unsigned ldst = lds_base + (unsigned)buf * 16384 + tid * 128;
#pragma unroll
    for (int j = 0; j < 8; ++j)
      async_b128_to_lds(ldst + j * 16, gsrc + j * 16);
  };

  issue_slice(0, 0);

  // ---- layer 0: Y0[16,256] = relu(bn(X[16,384] @ W0^T)) ----
  v8f acc[16];
#pragma unroll
  for (int nt = 0; nt < 16; ++nt)
#pragma unroll
    for (int j = 0; j < 8; ++j) acc[nt][j] = 0.0f;

  const _Float16* xrow = Xh + (size_t)(rowBase + ln) * INCH;
  v8h p0 = *(const v8h*)(xrow + 8 * hi);
  v8h p1 = *(const v8h*)(xrow + 16 + 8 * hi);
  v16h A = __builtin_shufflevector(p0, p1,
      0, 1, 2, 3, 4, 5, 6, 7, 8, 9, 10, 11, 12, 13, 14, 15);

  for (int ks = 0; ks < NK0; ++ks) {
    __syncthreads();                  // prior reads of buf[(ks+1)&1] finished
    if (ks + 1 < NK0) {
      issue_slice(ks + 1, (ks + 1) & 1);
      WAIT_ASYNC(8);                  // drain previous slice, keep new in flight
    } else {
      WAIT_ASYNC(0);
    }
    __syncthreads();                  // slice ks visible to all waves

    // prefetch next A fragment while wmma runs
    v16h Anext = A;
    if (ks + 1 < NK0) {
      const int k0n = (ks + 1) * 32;
      v8h q0 = *(const v8h*)(xrow + k0n + 8 * hi);
      v8h q1 = *(const v8h*)(xrow + k0n + 16 + 8 * hi);
      Anext = __builtin_shufflevector(q0, q1,
          0, 1, 2, 3, 4, 5, 6, 7, 8, 9, 10, 11, 12, 13, 14, 15);
    }

    // B fragments: explicitly rotated depth-2 software pipeline so the
    // scheduler keeps 2 LDS loads in flight (partial s_wait_dscnt, no drain).
    const _Float16* wb = sh + (size_t)(ks & 1) * 8192;
    v16h bA = *(const v16h*)(wb + (0 * 32 + lane) * 16);
    v16h bB = *(const v16h*)(wb + (1 * 32 + lane) * 16);
#pragma unroll
    for (int nt = 0; nt < 16; ++nt) {
      v16h bC = bA;
      if (nt + 2 < 16)
        bC = *(const v16h*)(wb + ((nt + 2) * 32 + lane) * 16);
      acc[nt] = __builtin_amdgcn_wmma_f32_16x16x32_f16(
          false, A, false, bA, (short)0, acc[nt], false, false);
      bA = bB;
      bB = bC;
    }
    A = Anext;
  }

  __syncthreads();   // all waves done with W buffers; smem becomes Y0

  // BN + ReLU, stage Y0 in LDS as f16 (row-major, padded stride)
#pragma unroll
  for (int nt = 0; nt < 16; ++nt) {
    const int o = nt * 16 + ln;
    const float s = s0g[o];
    const float t = t0g[o];
#pragma unroll
    for (int r = 0; r < 8; ++r) {
      float v = fmaf(acc[nt][r], s, t);
      v = v > 0.0f ? v : 0.0f;
      sh[((wave * 16) + (r + 8 * hi)) * Y0_STRIDE + o] = (_Float16)v;
    }
  }
  __syncthreads();

  // ---- layer 1: Y1[16,128] = relu(bn(Y0[16,256] @ W1^T)) ----
  v8f acc2[8];
#pragma unroll
  for (int nt = 0; nt < 8; ++nt)
#pragma unroll
    for (int j = 0; j < 8; ++j) acc2[nt][j] = 0.0f;

  const _Float16* yr = sh + (size_t)(wave * 16 + ln) * Y0_STRIDE;
  for (int ks = 0; ks < NK1; ++ks) {
    const int k0 = ks * 32;
    const v8h c0 = *(const v8h*)(yr + k0 + 8 * hi);
    const v8h c1 = *(const v8h*)(yr + k0 + 16 + 8 * hi);
    const v16h A2 = __builtin_shufflevector(c0, c1,
        0, 1, 2, 3, 4, 5, 6, 7, 8, 9, 10, 11, 12, 13, 14, 15);

    // rotated depth-2 pipeline over the 8 global B fragments
    const _Float16* wk = w1p + (size_t)ks * 4096;
    v16h bA = *(const v16h*)(wk + (0 * 32 + lane) * 16);
    v16h bB = *(const v16h*)(wk + (1 * 32 + lane) * 16);
#pragma unroll
    for (int nt = 0; nt < 8; ++nt) {
      v16h bC = bA;
      if (nt + 2 < 8)
        bC = *(const v16h*)(wk + ((nt + 2) * 32 + lane) * 16);
      acc2[nt] = __builtin_amdgcn_wmma_f32_16x16x32_f16(
          false, A2, false, bA, (short)0, acc2[nt], false, false);
      bA = bB;
      bB = bC;
    }
  }

  // BN + ReLU, store transposed: out[b, o, n]; lane's 8 values contiguous in n
  const int b    = rowBase >> 14;
  const int nloc = rowBase & (NPTS - 1);
#pragma unroll
  for (int nt = 0; nt < 8; ++nt) {
    const int o = nt * 16 + ln;
    const float s = s1g[o];
    const float t = t1g[o];
    f4 lo, hi4;
#pragma unroll
    for (int r = 0; r < 4; ++r) {
      float v = fmaf(acc2[nt][r], s, t);
      lo[r] = v > 0.0f ? v : 0.0f;
    }
#pragma unroll
    for (int r = 0; r < 4; ++r) {
      float v = fmaf(acc2[nt][r + 4], s, t);
      hi4[r] = v > 0.0f ? v : 0.0f;
    }
    float* op = out + ((size_t)b * OUT1 + o) * NPTS + nloc + 8 * hi;
    *(f4*)(op)     = lo;
    *(f4*)(op + 4) = hi4;
  }
}

// ---------------------------------------------------------------------------
extern "C" void kernel_launch(void* const* d_in, const int* in_sizes, int n_in,
                              void* d_out, int out_size, void* d_ws, size_t ws_size,
                              hipStream_t stream) {
  const float* xyz1    = (const float*)d_in[0];
  const float* xyz2    = (const float*)d_in[1];
  const float* points1 = (const float*)d_in[2];
  const float* points2 = (const float*)d_in[3];
  const float* w0  = (const float*)d_in[4];
  const float* b0  = (const float*)d_in[5];
  const float* g0  = (const float*)d_in[6];
  const float* bt0 = (const float*)d_in[7];
  const float* rm0 = (const float*)d_in[8];
  const float* rv0 = (const float*)d_in[9];
  const float* w1  = (const float*)d_in[10];
  const float* b1  = (const float*)d_in[11];
  const float* g1  = (const float*)d_in[12];
  const float* bt1 = (const float*)d_in[13];
  const float* rm1 = (const float*)d_in[14];
  const float* rv1 = (const float*)d_in[15];

  char* ws = (char*)d_ws;
  _Float16* Xh  = (_Float16*)(ws + OFF_XH);
  _Float16* w0p = (_Float16*)(ws + OFF_W0P);
  _Float16* w1p = (_Float16*)(ws + OFF_W1P);
  float* s0 = (float*)(ws + OFF_S0);
  float* t0 = (float*)(ws + OFF_T0);
  float* s1 = (float*)(ws + OFF_S1);
  float* t1 = (float*)(ws + OFF_T1);

  // 0) fold BN, convert + fragment-pack weights to f16
  fp_prep_kernel<<<(OUT0 * INCH + 255) / 256, 256, 0, stream>>>(
      w0, b0, g0, bt0, rm0, rv0, w1, b1, g1, bt1, rm1, rv1,
      w0p, w1p, s0, t0, s1, t1);

  // 1) 3-NN interpolation + f16 feature assembly
  fp_interp_kernel<<<(BATCH * NPTS) / 256, 256, 0, stream>>>(
      xyz1, xyz2, points1, points2, Xh);

  // 2) fused WMMA MLP with async-LDS weight staging, transposed output
  fp_mlp_kernel<<<(BATCH * NPTS) / 64, 128, 0, stream>>>(
      Xh, w0p, w1p, s0, t0, s1, t1, (float*)d_out);
}
